// SkipIndRNN_26955214749946
// MI455X (gfx1250) — matrix operand order repaired and checked by reading, not compiled, not run on hardware
//
#include <hip/hip_runtime.h>

typedef __attribute__((ext_vector_type(16))) __bf16 v16bf;
typedef __attribute__((ext_vector_type(8)))  __bf16 v8bf;
typedef __attribute__((ext_vector_type(8)))  float  v8f;

#define TT 512
#define BB 64
#define II 256
#define HH 512
#define MM (TT*BB)

// fp32 -> bf16 prepack (vectorized x4).  n4 = count/4.
__global__ __launch_bounds__(256)
void cvt_bf16_kernel(const float* __restrict__ src, __bf16* __restrict__ dst, int n4) {
  const int i = blockIdx.x * 256 + threadIdx.x;
  if (i < n4) {
    const float4 v = ((const float4*)src)[i];
    union { __bf16 b[4]; uint2 u; } t;
    t.b[0] = (__bf16)v.x; t.b[1] = (__bf16)v.y;
    t.b[2] = (__bf16)v.z; t.b[3] = (__bf16)v.w;
    ((uint2*)dst)[i] = t.u;
  }
}

// out[M,N] = A[M,K] @ W[N,K]^T + bias[N], N = HH.
// Block: 256 threads = 8 waves; covers a 32-row strip of A and all N=512.
// Wave w: n in [w*64,(w+1)*64) as four 16x16 n-tiles x two 16-row m-tiles
// -> 8 accumulators, every B load feeds two WMMAs.
// A strip (contiguous 32*K fp32) staged once to LDS as bf16; B read as
// prepacked bf16 straight from L2 (W is L2-resident).
template <int K>
__global__ __launch_bounds__(256)
void gemm_bias_kernel(const float* __restrict__ A,
                      const __bf16* __restrict__ Wbf,
                      const float* __restrict__ bias,
                      float* __restrict__ out) {
  __shared__ __bf16 ldsA[32 * K];
  const int tid  = threadIdx.x;
  const int row0 = blockIdx.x * 32;

  // Stage: rows row0..row0+31 are contiguous (row-major, full rows).
  const float4* Af = (const float4*)(A + (size_t)row0 * K);
  #pragma unroll
  for (int i = tid; i < 8 * K; i += 256) {   // 32*K/4 float4s
    const float4 v = Af[i];
    union { __bf16 b[4]; uint2 u; } t;
    t.b[0] = (__bf16)v.x; t.b[1] = (__bf16)v.y;
    t.b[2] = (__bf16)v.z; t.b[3] = (__bf16)v.w;
    *(uint2*)&ldsA[i * 4] = t.u;
  }
  __syncthreads();

  const int lane  = tid & 31;
  const int wave  = tid >> 5;
  const int half  = lane >> 4;   // 16-lane half of the wave
  const int l16   = lane & 15;
  const int nbase = wave * 64;

  v8f acc[2][4];
  #pragma unroll
  for (int mt = 0; mt < 2; ++mt)
    #pragma unroll
    for (int j = 0; j < 4; ++j) acc[mt][j] = v8f{};

  // Unroll capped at 2: keeps ~2 k-steps of A/B tiles in flight (enough to
  // cover L2 latency with 8 waves/block) while staying under 256 VGPRs so
  // codegen avoids s_set_vgpr_msb churn in the hot loop.
  #pragma unroll 2
  for (int k0 = 0; k0 < K; k0 += 32) {
    // A 16x32 bf16 per-lane layout (matches B):
    //   half 0: K = k0+{0..15}; half 1: K = k0+{16..31}
    v16bf a[2];
    #pragma unroll
    for (int mt = 0; mt < 2; ++mt) {
      const int rbase = (mt * 16 + l16) * K;
      const v8bf alo = *(const v8bf*)&ldsA[rbase + k0 + half * 16];
      const v8bf ahi = *(const v8bf*)&ldsA[rbase + k0 + half * 16 + 8];
      a[mt] = __builtin_shufflevector(alo, ahi,
          0,1,2,3,4,5,6,7,8,9,10,11,12,13,14,15);
    }
    #pragma unroll
    for (int j = 0; j < 4; ++j) {
      // B 32x16: half 0 holds K=k0..k0+15 of column n, half 1 K=k0+16..k0+31.
      // B[k][n] = W[n][k]; prepacked bf16 row-major [N,K] -> one 32B load.
      const int n = nbase + j * 16 + l16;
      const v16bf b = *(const v16bf*)&Wbf[(size_t)n * K + k0 + half * 16];
      acc[0][j] = __builtin_amdgcn_wmma_f32_16x16x32_bf16(
          false, a[0], false, b, (short)0, acc[0][j], false, false);
      acc[1][j] = __builtin_amdgcn_wmma_f32_16x16x32_bf16(
          false, a[1], false, b, (short)0, acc[1][j], false, false);
    }
  }

  // C/D layout: VGPR r -> lanes 0-15: M=r, lanes 16-31: M=8+r.
  #pragma unroll
  for (int j = 0; j < 4; ++j) {
    const int n   = nbase + j * 16 + l16;
    const float bv = bias[n];
    #pragma unroll
    for (int mt = 0; mt < 2; ++mt) {
      #pragma unroll
      for (int r = 0; r < 8; ++r) {
        const int m = mt * 16 + r + half * 8;
        out[(size_t)(row0 + m) * HH + n] = acc[mt][j][r] + bv;
      }
    }
  }
}

// Skip-IndRNN scan: recurrence is elementwise in (b,h) -> one thread per pair,
// serial over t.  preg holds `pre` on entry and is overwritten in place with
// the gate output g (each element read exactly once before its write).
__global__ __launch_bounds__(256)
void scan_kernel(float* __restrict__ preg,
                 float* __restrict__ hout,
                 const float* __restrict__ h0,
                 const float* __restrict__ w_hh,
                 const float* __restrict__ w_uh,
                 const float* __restrict__ b_uh) {
  const int idx = blockIdx.x * 256 + threadIdx.x;
  const int BH  = BB * HH;
  if (idx >= BH) return;
  const int h = idx & (HH - 1);

  float w = w_hh[h];
  w = fminf(fmaxf(w, -1000.0f), 1000.0f);   // HIDDEN_MAX_ABS clamp
  const float wu = w_uh[h];
  const float bu = b_uh[0];

  float hx  = h0[h];
  float up  = 1.0f;
  float cum = 0.0f;

  for (int t = 0; t < TT; ++t) {
    const size_t off = (size_t)t * BH + idx;
    const float p      = preg[off];
    const float new_hx = fmaxf(fmaf(w, hx, p), 0.0f);                    // relu
    const float tilde  = 1.0f / (1.0f + __expf(-fmaf(new_hx, wu, bu)));  // sigmoid
    const float c      = cum + fminf(up, 1.0f - cum);
    const float g      = rintf(c);          // round-half-even == jnp.round
    const float nh     = g * new_hx + (1.0f - g) * hx;
    up  = g * tilde + (1.0f - g) * up;
    cum = (1.0f - g) * c;
    hx  = nh;
    hout[off] = nh;
    preg[off] = g;
  }
}

extern "C" void kernel_launch(void* const* d_in, const int* in_sizes, int n_in,
                              void* d_out, int out_size, void* d_ws, size_t ws_size,
                              hipStream_t stream) {
  const float* x    = (const float*)d_in[0];  // [T,B,I]
  const float* h0   = (const float*)d_in[1];  // [L,H]
  const float* Wih0 = (const float*)d_in[2];  // [H,I]
  const float* Wih1 = (const float*)d_in[3];  // [H,H]
  const float* bih  = (const float*)d_in[4];  // [L,H]
  const float* whh  = (const float*)d_in[5];  // [L,H]
  const float* wuh  = (const float*)d_in[6];  // [L,H]
  const float* buh  = (const float*)d_in[7];  // [L]

  float* base = (float*)d_out;
  const size_t plane = (size_t)TT * BB * HH;
  float* o0 = base;                 // x_out[0]
  float* o1 = base + plane;         // x_out[1]
  float* g0 = base + 2 * plane;     // gates[0]  (also scratch for pre0)
  float* g1 = base + 3 * plane;     // gates[1]  (also scratch for pre1)

  // Prepacked bf16 weights live at the head of the x_out[1] plane: it is only
  // consumed before scan-1 runs, and scan-1 (the last kernel) fully rewrites it.
  __bf16* Wbf0 = (__bf16*)o1;                 // HH*II bf16  (256 KB)
  __bf16* Wbf1 = Wbf0 + (size_t)HH * II;      // HH*HH bf16  (512 KB)

  const dim3 blk(256);
  cvt_bf16_kernel<<<dim3((HH * II / 4 + 255) / 256), blk, 0, stream>>>(Wih0, Wbf0, HH * II / 4);
  cvt_bf16_kernel<<<dim3((HH * HH / 4 + 255) / 256), blk, 0, stream>>>(Wih1, Wbf1, HH * HH / 4);

  const dim3 gemm_grid(MM / 32);
  const dim3 scan_grid((BB * HH) / 256);

  // Layer 0
  gemm_bias_kernel<II><<<gemm_grid, blk, 0, stream>>>(x, Wbf0, bih, g0);
  scan_kernel<<<scan_grid, blk, 0, stream>>>(g0, o0, h0, whh, wuh, buh);
  // Layer 1 (input is o0, K = H)
  gemm_bias_kernel<HH><<<gemm_grid, blk, 0, stream>>>(o0, Wbf1, bih + HH, g1);
  scan_kernel<<<scan_grid, blk, 0, stream>>>(g1, o1, h0 + HH, whh + HH, wuh + HH, buh + 1);
}
